// MultiheadAttention_41162966565556
// MI455X (gfx1250) — matrix operand order repaired and checked by reading, not compiled, not run on hardware
//
#include <hip/hip_runtime.h>

// ---------------------------------------------------------------------------
// CDNA5 (gfx1250) multi-head attention, bf16 WMMA with fp32 accumulation.
// wave32; WMMA shape 16x16x32 bf16 -> f32.
// ---------------------------------------------------------------------------

typedef __attribute__((ext_vector_type(16))) __bf16        v16bf;
typedef __attribute__((ext_vector_type(8)))  float         v8f;
typedef __attribute__((ext_vector_type(8)))  unsigned short ush8;

#define DEV __device__ __forceinline__

DEV unsigned short f2bf(float f) {
  unsigned int u = __builtin_bit_cast(unsigned int, f);
  u += 0x7FFFu + ((u >> 16) & 1u);          // round-to-nearest-even
  return (unsigned short)(u >> 16);
}

union Frag {
  v16bf         bf;     // 16 bf16 = 8 VGPRs
  ush8          u8[2];  // two 16-byte chunks
  unsigned short s[16];
};

DEV v8f wmma_bf16(const Frag& a, const Frag& b, v8f c) {
  // (neg_a, A, neg_b, B, c_mod, C, reuse_a, reuse_b)
  return __builtin_amdgcn_wmma_f32_16x16x32_bf16(false, a.bf, false, b.bf,
                                                 (short)0, c, false, false);
}

// Fragment layout assumptions (CDNA5 ISA):
//  A (16x32, 16-bit): lane m = lane&15, half = lane>>4;
//    chunk0: K = half*8 .. +7   (VGPR0..3)
//    chunk1: K = 16+half*8 .. +7 (VGPR4..7)
//  B (32x16, 16-bit): lane n = lane&15, half = lane>>4;
//    same K order as A: K = half*8 .. +7, then 16+half*8 .. +7
//  C/D (16x16 f32): element r of v8f -> m = r + 8*half, n = lane&15.

// ---------------------------------------------------------------------------
// Tiled GEMM: out[M,1024] = X[M,1024] @ W[1024,1024] + bias
// block = 128 threads (4 waves), tile 64x64, BK = 32.
// TIN = float (convert to bf16) or unsigned short (bf16 passthrough).
// OUT_F32: write fp32; TRANS_V: write bf16 into [B][H][D][S] transposed layout.
// ---------------------------------------------------------------------------
template <typename TIN, bool OUT_F32, bool TRANS_V>
__launch_bounds__(128)
__global__ void gemm_bias_kernel(const TIN* __restrict__ X,
                                 const float* __restrict__ W,
                                 const float* __restrict__ bias,
                                 void* __restrict__ outp) {
  constexpr int N = 1024, K = 1024;
  constexpr int LDA = 40, LDB = 40;            // padded ushort strides
  __shared__ unsigned short As[64 * LDA];
  __shared__ unsigned short Bs[64 * LDB];

  const int tid  = threadIdx.x;
  const int wave = tid >> 5, lane = tid & 31;
  const int half = lane >> 4, l16 = lane & 15;
  const int m0 = blockIdx.x * 64, n0 = blockIdx.y * 64;
  const int wm = (wave >> 1) * 32, wn = (wave & 1) * 32;

  v8f c[2][2] = {};

  for (int k0 = 0; k0 < K; k0 += 32) {
    __syncthreads();
    // ---- stage A tile (64 rows x 32 k) as bf16 ----
    if constexpr (sizeof(TIN) == 4) {
      for (int it = tid; it < 512; it += 128) {         // 64*8 float4
        const int r = it >> 3, cc = (it & 7) * 4;
        const float4 v = *(const float4*)((const float*)X + (size_t)(m0 + r) * K + k0 + cc);
        unsigned short* d = &As[r * LDA + cc];
        d[0] = f2bf(v.x); d[1] = f2bf(v.y); d[2] = f2bf(v.z); d[3] = f2bf(v.w);
      }
    } else {
      for (int it = tid; it < 256; it += 128) {         // 64*4 ush8
        const int r = it >> 2, cc = (it & 3) * 8;
        *(ush8*)&As[r * LDA + cc] =
            *(const ush8*)((const unsigned short*)X + (size_t)(m0 + r) * K + k0 + cc);
      }
    }
    // ---- stage W tile transposed: Bs[n][k] (32 k-rows x 64 n-cols) ----
    for (int it = tid; it < 512; it += 128) {           // 32*16 float4
      const int kk = it >> 4, ng = (it & 15) * 4;
      const float4 v = *(const float4*)(W + (size_t)(k0 + kk) * N + n0 + ng);
      Bs[(ng + 0) * LDB + kk] = f2bf(v.x);
      Bs[(ng + 1) * LDB + kk] = f2bf(v.y);
      Bs[(ng + 2) * LDB + kk] = f2bf(v.z);
      Bs[(ng + 3) * LDB + kk] = f2bf(v.w);
    }
    __syncthreads();

    Frag a[2], b[2];
#pragma unroll
    for (int i = 0; i < 2; ++i) {
      const unsigned short* ar = &As[(wm + i * 16 + l16) * LDA];
      a[i].u8[0] = *(const ush8*)(ar + half * 8);
      a[i].u8[1] = *(const ush8*)(ar + 16 + half * 8);
      const unsigned short* br = &Bs[(wn + i * 16 + l16) * LDB];
      b[i].u8[0] = *(const ush8*)(br + half * 8);
      b[i].u8[1] = *(const ush8*)(br + 16 + half * 8);
    }
#pragma unroll
    for (int i = 0; i < 2; ++i)
#pragma unroll
      for (int j = 0; j < 2; ++j)
        c[i][j] = wmma_bf16(a[i], b[j], c[i][j]);
  }

  // ---- epilogue: bias + store ----
#pragma unroll
  for (int i = 0; i < 2; ++i) {
#pragma unroll
    for (int j = 0; j < 2; ++j) {
      const int n  = n0 + wn + j * 16 + l16;
      const float bv = bias[n];
#pragma unroll
      for (int r = 0; r < 8; ++r) {
        const int   m   = m0 + wm + i * 16 + half * 8 + r;
        const float val = c[i][j][r] + bv;
        if constexpr (OUT_F32) {
          ((float*)outp)[(size_t)m * N + n] = val;
        } else if constexpr (TRANS_V) {
          const int h = n >> 6, d = n & 63;           // n = h*64 + d
          const int bb = m >> 11, s = m & 2047;       // m = bb*2048 + s
          ((unsigned short*)outp)[(((size_t)bb * 16 + h) * 64 + d) * 2048 + s] = f2bf(val);
        } else {
          ((unsigned short*)outp)[(size_t)m * N + n] = f2bf(val);
        }
      }
    }
  }
}

// ---------------------------------------------------------------------------
// Attention: one block per (b, h, 16-query tile). 128 threads (4 waves).
// Whole 16x2048 score block lives in dynamic LDS (CDNA5: 320KB/WGP).
// ---------------------------------------------------------------------------
__launch_bounds__(128)
__global__ void attention_kernel(const unsigned short* __restrict__ Qb,  // [B,S,E] bf16
                                 const unsigned short* __restrict__ Kb,  // [B,S,E] bf16
                                 const unsigned short* __restrict__ Vt,  // [B,H,D,S] bf16
                                 float* __restrict__ attn,               // [B,H,S,S] f32
                                 unsigned short* __restrict__ Wt) {      // [B,S,E] bf16
  constexpr int S = 2048, E = 1024, D = 64, H = 16;
  constexpr int SROW = S + 4;                 // padded f32 stride (bank spread)
  extern __shared__ float scores[];           // 16*SROW + 128 floats
  float* red = scores + 16 * SROW;

  const int tid  = threadIdx.x;
  const int wave = tid >> 5, lane = tid & 31;
  const int half = lane >> 4, l16 = lane & 15;
  const int bid = blockIdx.x;
  const int qt  = bid & 127;
  const int h   = (bid >> 7) & 15;
  const int b   = bid >> 11;

  const unsigned short* Qh = Qb + (size_t)b * S * E + h * D;
  const unsigned short* Kh = Kb + (size_t)b * S * E + h * D;
  const unsigned short* Vh = Vt + ((size_t)(b * H + h)) * D * S;
  float* attn_tile = attn + (((size_t)(b * H + h)) * S + (size_t)qt * 16) * S;

  // Q fragments (K = 0..63 in two chunks) held in registers for all of phase 1
  Frag aq0, aq1;
  {
    const unsigned short* qr = Qh + (size_t)(qt * 16 + l16) * E;
    aq0.u8[0] = *(const ush8*)(qr + half * 8);
    aq0.u8[1] = *(const ush8*)(qr + 16 + half * 8);
    aq1.u8[0] = *(const ush8*)(qr + 32 + half * 8);
    aq1.u8[1] = *(const ush8*)(qr + 48 + half * 8);
  }

  // ---- phase 1: scores = Q K^T * (1/sqrt(64)) ----
  for (int kt = wave; kt < S / 16; kt += 4) {
    Frag b0, b1;
    const unsigned short* kr = Kh + (size_t)(kt * 16 + l16) * E;
    b0.u8[0] = *(const ush8*)(kr + half * 8);
    b0.u8[1] = *(const ush8*)(kr + 16 + half * 8);
    b1.u8[0] = *(const ush8*)(kr + 32 + half * 8);
    b1.u8[1] = *(const ush8*)(kr + 48 + half * 8);
    v8f c = {};
    c = wmma_bf16(aq0, b0, c);
    c = wmma_bf16(aq1, b1, c);
#pragma unroll
    for (int r = 0; r < 8; ++r)
      scores[(half * 8 + r) * SROW + kt * 16 + l16] = c[r] * 0.125f;
  }
  __syncthreads();

  // ---- phase 2: fp32 softmax over each row (8 threads per row) ----
  {
    const int row = tid >> 3, sub = tid & 7;
    float* srow = scores + row * SROW;
    float lmax = -3.0e38f;
    for (int cc = sub; cc < S; cc += 8) lmax = fmaxf(lmax, srow[cc]);
    red[tid] = lmax;
    __syncthreads();
    float mx = red[row * 8];
#pragma unroll
    for (int k = 1; k < 8; ++k) mx = fmaxf(mx, red[row * 8 + k]);
    float lsum = 0.f;
    for (int cc = sub; cc < S; cc += 8) {
      const float e = __expf(srow[cc] - mx);
      srow[cc] = e;
      lsum += e;
    }
    __syncthreads();
    red[tid] = lsum;
    __syncthreads();
    float sum = 0.f;
#pragma unroll
    for (int k = 0; k < 8; ++k) sum += red[row * 8 + k];
    const float inv = 1.0f / sum;
    for (int cc = sub; cc < S; cc += 8) srow[cc] *= inv;
  }
  __syncthreads();

  // ---- write attn_weights (dominant HBM traffic; fully coalesced) ----
  for (int idx = tid; idx < 16 * S; idx += 128) {
    const int m = idx >> 11, cc = idx & (S - 1);
    attn_tile[(size_t)m * S + cc] = scores[m * SROW + cc];
  }

  // ---- phase 3: weighted = P @ V, one 16-dim output slice per wave ----
  v8f acc = {};
  for (int kt = 0; kt < S / 32; ++kt) {
    Frag a, bv;
    const float* sr = scores + l16 * SROW + kt * 32;
#pragma unroll
    for (int j = 0; j < 8; ++j) {
      a.s[j]     = f2bf(sr[half * 8 + j]);
      a.s[8 + j] = f2bf(sr[16 + half * 8 + j]);
    }
    const unsigned short* vr = Vh + (size_t)(wave * 16 + l16) * S + kt * 32 + half * 8;
    bv.u8[0] = *(const ush8*)(vr);
    bv.u8[1] = *(const ush8*)(vr + 16);
    acc = wmma_bf16(a, bv, acc);
  }
  {
    unsigned short* wrow =
        Wt + ((size_t)(b * S + qt * 16)) * E + h * D + wave * 16 + l16;
#pragma unroll
    for (int r = 0; r < 8; ++r)
      wrow[(size_t)(half * 8 + r) * E] = f2bf(acc[r]);
  }
}

// ---------------------------------------------------------------------------
extern "C" void kernel_launch(void* const* d_in, const int* in_sizes, int n_in,
                              void* d_out, int out_size, void* d_ws, size_t ws_size,
                              hipStream_t stream) {
  (void)in_sizes; (void)n_in; (void)out_size; (void)ws_size;
  const float* query = (const float*)d_in[0];
  const float* key   = (const float*)d_in[1];
  const float* value = (const float*)d_in[2];
  const float* Wq = (const float*)d_in[3];
  const float* bq = (const float*)d_in[4];
  const float* Wk = (const float*)d_in[5];
  const float* bk = (const float*)d_in[6];
  const float* Wv = (const float*)d_in[7];
  const float* bv = (const float*)d_in[8];
  const float* Wo = (const float*)d_in[9];
  const float* bo = (const float*)d_in[10];

  constexpr int Bb = 2, S = 2048, E = 1024, H = 16;
  constexpr size_t MTOT = (size_t)Bb * S;           // 4096 rows

  float* out = (float*)d_out;                       // [B,S,E] fp32
  float* attn = out + MTOT * E;                     // [B,H,S,S] fp32

  // workspace: bf16 Q, K, V^T, weighted  (4 x 8 MB = 32 MB)
  unsigned short* Qb = (unsigned short*)d_ws;
  unsigned short* Kb = Qb + MTOT * E;
  unsigned short* Vt = Kb + MTOT * E;
  unsigned short* Wt = Vt + MTOT * E;

  const dim3 grid(MTOT / 64, E / 64), blk(128);
  gemm_bias_kernel<float, false, false><<<grid, blk, 0, stream>>>(query, Wq, bq, Qb);
  gemm_bias_kernel<float, false, false><<<grid, blk, 0, stream>>>(key,   Wk, bk, Kb);
  gemm_bias_kernel<float, false, true ><<<grid, blk, 0, stream>>>(value, Wv, bv, Vt);

  const size_t lds = (size_t)(16 * (S + 4) + 128) * sizeof(float);  // ~128.5 KB
  attention_kernel<<<dim3(Bb * H * (S / 16)), blk, lds, stream>>>(Qb, Kb, Vt, attn, Wt);

  gemm_bias_kernel<unsigned short, true, false><<<grid, blk, 0, stream>>>(Wt, Wo, bo, out);
}
